// EMACluster_77309411658
// MI455X (gfx1250) — compile-verified, matrix operand order checked by reading, not run on hardware
//
#include <hip/hip_runtime.h>

typedef __attribute__((ext_vector_type(16))) __bf16 v16bf;
typedef __attribute__((ext_vector_type(8)))  float  v8f;

#define N_E      1024
#define E_DIM    256
#define HWSZ     4096        // 64*64
#define ZQ_ELEMS 33554432    // 32*256*64*64

union Frag { v16bf v; uint4 q[2]; };

__device__ __forceinline__ unsigned int pack_bf16x2(float a, float b) {
  unsigned int ua = __float_as_uint(a);
  unsigned int ub = __float_as_uint(b);
  ua = (ua + 0x7FFFu + ((ua >> 16) & 1u)) >> 16;   // RNE f32 -> bf16
  ub = (ub + 0x7FFFu + ((ub >> 16) & 1u)) >> 16;
  return ua | (ub << 16);
}

// ---------------------------------------------------------------------------
// Kernel 1: emb (1024x256 f32) -> bf16 rows + per-code squared norms.
// Also zero-initializes the loss accumulator (stream-ordered, capture-safe).
// ---------------------------------------------------------------------------
__global__ void vq_prep(const float* __restrict__ emb,
                        unsigned short* __restrict__ embB,
                        float* __restrict__ normE,
                        float* __restrict__ lossAcc) {
  int j = blockIdx.x * blockDim.x + threadIdx.x;
  if (j == 0) *lossAcc = 0.0f;
  if (j >= N_E) return;
  const float* er = emb + (size_t)j * E_DIM;
  unsigned int* ob = (unsigned int*)(embB + (size_t)j * E_DIM);
  float s = 0.0f;
#pragma unroll 4
  for (int c = 0; c < E_DIM; c += 2) {
    float a = er[c], b = er[c + 1];
    s += a * a + b * b;
    ob[c >> 1] = pack_bf16x2(a, b);
  }
  normE[j] = s;
}

// ---------------------------------------------------------------------------
// Kernel 2: WMMA distance GEMM + argmin.
// Block = 256 threads = 8 waves; block tile = 128 pixels; wave tile = 16 pixels.
// All 8 B fragments of an N-tile are loaded into registers before the WMMA
// sequence (loads issue as clauses, waits staggered), and the 8 WMMAs are
// split across two accumulators to halve the RAW dependency chain.
// Argmin uses score = |e|^2 - 2*dot (the |z|^2 term is row-constant).
// ---------------------------------------------------------------------------
__global__ __launch_bounds__(256, 2)
void vq_argmin(const float* __restrict__ z,
               const unsigned short* __restrict__ embB,
               const float* __restrict__ normE,
               int* __restrict__ idxI,
               float* __restrict__ idxF) {
  __shared__ unsigned int ldsA[16384];   // 128 rows x 256 bf16 (64 KB), swizzled

  const int tid  = threadIdx.x;
  const int pix  = tid & 127;
  const int hh   = tid >> 7;             // which half of the block
  const int bimg = blockIdx.x >> 5;      // image index (4096 % 128 == 0)
  const int hw0  = (blockIdx.x & 31) * 128;
  const float* zb = z + ((size_t)bimg * E_DIM) * HWSZ + hw0;

  // Stage A tile: 128 pixels x 256 channels, f32 -> bf16x2, coalesced loads.
  for (int it = 0; it < 64; ++it) {
    int cp = it * 2 + hh;                // channel-pair index 0..127
    int c0 = cp * 2;
    float f0 = zb[(size_t)c0 * HWSZ + pix];
    float f1 = zb[(size_t)(c0 + 1) * HWSZ + pix];
    ldsA[pix * 128 + (((cp >> 2) ^ (pix & 15)) << 2) + (cp & 3)] =
        pack_bf16x2(f0, f1);
  }
  __syncthreads();

  const int wave = tid >> 5;
  const int lane = tid & 31;
  const int row  = wave * 16 + (lane & 15);
  const int ks   = (lane < 16) ? 0 : 8;  // ISA A/B 16-bit fragment K split

  // Preload all A fragments for this wave's 16 pixels (reused for all 64 N-tiles).
  Frag af[8];
#pragma unroll
  for (int kc = 0; kc < 8; ++kc) {
    int ca = (kc * 32 + ks) >> 3;        // 16B chunk index (8 bf16 per chunk)
    int cb = (kc * 32 + ks + 16) >> 3;
    af[kc].q[0] = *(const uint4*)&ldsA[row * 128 + ((ca ^ (row & 15)) << 2)];
    af[kc].q[1] = *(const uint4*)&ldsA[row * 128 + ((cb ^ (row & 15)) << 2)];
  }

  float bestV[8];
  int   bestI[8];
#pragma unroll
  for (int r = 0; r < 8; ++r) { bestV[r] = 3.0e38f; bestI[r] = 0; }

  for (int n0 = 0; n0 < N_E; n0 += 16) {
    int col = n0 + (lane & 15);
    const unsigned short* eb = embB + (size_t)col * E_DIM;
    if (n0 + 16 < N_E) __builtin_prefetch(eb + E_DIM, 0, 0);

    // Issue all 16 B-fragment loads first so VMEM latency overlaps the WMMAs.
    Frag bf[8];
#pragma unroll
    for (int kc = 0; kc < 8; ++kc) {
      bf[kc].q[0] = *(const uint4*)(eb + kc * 32 + ks);
      bf[kc].q[1] = *(const uint4*)(eb + kc * 32 + ks + 16);
    }

    // Two accumulators: halves the WMMA->WMMA RAW chain (two ops in flight).
    v8f acc0 = {};
    v8f acc1 = {};
#pragma unroll
    for (int kc = 0; kc < 8; kc += 2) {
      acc0 = __builtin_amdgcn_wmma_f32_16x16x32_bf16(
          false, af[kc].v,     false, bf[kc].v,     (short)0, acc0, false, false);
      acc1 = __builtin_amdgcn_wmma_f32_16x16x32_bf16(
          false, af[kc + 1].v, false, bf[kc + 1].v, (short)0, acc1, false, false);
    }

    float ne = normE[col];
    // C layout: lane holds column `col`, VGPR r holds row r (lanes 0-15) or
    // r+8 (lanes 16-31). Track per-lane best over the 64 columns this lane sees.
#pragma unroll
    for (int r = 0; r < 8; ++r) {
      float d = ne - 2.0f * (acc0[r] + acc1[r]);
      if (d < bestV[r]) { bestV[r] = d; bestI[r] = col; }
    }
  }

  // One butterfly reduction per row across the 16 columns in each half-wave.
#pragma unroll
  for (int r = 0; r < 8; ++r) {
    float v = bestV[r]; int bi = bestI[r];
#pragma unroll
    for (int m = 8; m >= 1; m >>= 1) {
      float ov = __shfl_xor(v, m, 32);
      int   oi = __shfl_xor(bi, m, 32);
      if (ov < v || (ov == v && oi < bi)) { v = ov; bi = oi; }
    }
    bestV[r] = v; bestI[r] = bi;
  }

  size_t waveBase = (size_t)blockIdx.x * 128 + (size_t)wave * 16;
  if (lane == 0) {
#pragma unroll
    for (int r = 0; r < 8; ++r) {        // rows 0..7
      idxI[waveBase + r] = bestI[r];
      idxF[waveBase + r] = (float)bestI[r];
    }
  } else if (lane == 16) {
#pragma unroll
    for (int r = 0; r < 8; ++r) {        // rows 8..15
      idxI[waveBase + 8 + r] = bestI[r];
      idxF[waveBase + 8 + r] = (float)bestI[r];
    }
  }
}

// ---------------------------------------------------------------------------
// Kernel 3: gather z_q = emb[idx] (exact fp32) + MSE partial sums.
// Coalesced z read / z_q write per channel plane; emb rows hit L2.
// ---------------------------------------------------------------------------
__global__ __launch_bounds__(256)
void vq_gather_loss(const float* __restrict__ z,
                    const float* __restrict__ emb,
                    const int* __restrict__ idxI,
                    float* __restrict__ zq,
                    float* __restrict__ lossAcc) {
  __shared__ float red[256];
  const int tid  = threadIdx.x;
  const int pix  = tid & 127;
  const int hh   = tid >> 7;
  const int bimg = blockIdx.x >> 5;
  const int hw0  = (blockIdx.x & 31) * 128;
  const size_t base = ((size_t)bimg * E_DIM) * HWSZ + hw0 + pix;

  int code = idxI[(size_t)blockIdx.x * 128 + pix];
  const float* er = emb + (size_t)code * E_DIM;

  float s = 0.0f;
  for (int it = 0; it < 128; ++it) {
    int c = it * 2 + hh;
    float e  = er[c];
    size_t off = base + (size_t)c * HWSZ;
    float zv = z[off];
    zq[off] = e;
    float d = e - zv;
    s += d * d;
  }

  red[tid] = s;
  __syncthreads();
  for (int st = 128; st > 0; st >>= 1) {
    if (tid < st) red[tid] += red[tid + st];
    __syncthreads();
  }
  if (tid == 0) atomicAdd(lossAcc, red[0]);
}

__global__ void vq_finalize(const float* __restrict__ lossAcc,
                            float* __restrict__ out) {
  *out = *lossAcc * (1.0f / (float)ZQ_ELEMS);
}

// ---------------------------------------------------------------------------
extern "C" void kernel_launch(void* const* d_in, const int* in_sizes, int n_in,
                              void* d_out, int out_size, void* d_ws, size_t ws_size,
                              hipStream_t stream) {
  const float* z   = (const float*)d_in[0];   // (32, 256, 64, 64) f32
  const float* emb = (const float*)d_in[1];   // (1024, 256) f32

  float* out  = (float*)d_out;
  float* zq   = out;                          // 33,554,432 f32
  float* loss = out + ZQ_ELEMS;               // 1 f32
  float* idxF = out + ZQ_ELEMS + 1;           // 131,072 indices (as f32)

  char* ws = (char*)d_ws;
  float*          lossAcc = (float*)ws;                         // 4 B (+pad)
  float*          normE   = (float*)(ws + 256);                 // 4 KB
  unsigned short* embB    = (unsigned short*)(ws + 256 + 4096); // 512 KB bf16
  int*            idxI    = (int*)(ws + 256 + 4096 +
                                   (size_t)N_E * E_DIM * 2);    // 512 KB

  vq_prep       <<<4,    256, 0, stream>>>(emb, embB, normE, lossAcc);
  vq_argmin     <<<1024, 256, 0, stream>>>(z, embB, normE, idxI, idxF);
  vq_gather_loss<<<1024, 256, 0, stream>>>(z, emb, idxI, zq, lossAcc);
  vq_finalize   <<<1,    1,   0, stream>>>(lossAcc, loss);
}
